// AreaRNN_901943132268
// MI455X (gfx1250) — compile-verified, matrix operand order checked by reading, not run on hardware
//
#include <hip/hip_runtime.h>

#define NN 16384
#define KACT 50
#define MAXC 512
#define NBINS 1024

typedef __attribute__((ext_vector_type(16))) __bf16 v16bf;
typedef __attribute__((ext_vector_type(8)))  float  v8f;

union V16 { v16bf v; __bf16 e[16]; };

struct WS {
  int   cnt[2];
  int   pad[2];
  int   cols_x[MAXC];
  int   cols_y[MAXC];
  float yv[NN];
  int   hist[NBINS];
  int   thresh;
  int   ntake;
};

// ---------------- init: zero counters + histogram ----------------
__global__ void k_init(WS* ws) {
  int t = threadIdx.x;
  if (t < 2) ws->cnt[t] = 0;
  if (t < NBINS) ws->hist[t] = 0;
}

// ---------------- compact active indices of x / y_latent ----------------
__global__ void k_compact(const float* __restrict__ x,
                          const float* __restrict__ yl, WS* ws) {
  int i = blockIdx.x * blockDim.x + threadIdx.x;
  if (i >= NN) return;
  if (x[i] != 0.0f) {
    int p = atomicAdd(&ws->cnt[0], 1);
    if (p < MAXC) ws->cols_x[p] = i;
  }
  if (yl[i] != 0.0f) {
    int p = atomicAdd(&ws->cnt[1], 1);
    if (p < MAXC) ws->cols_y[p] = i;
  }
}

// ---------------- sparse-column matvec via WMMA row-sums ----------------
// Each wave computes 16 rows. A tile = 16 rows x 32 active columns (bf16,
// values are exactly 0/1), B = all-ones 32x16, f32 accumulate -> exact
// integer row sums. Column list is padded with index 0 so the 16 gathers per
// lane are unconditional (full MLP); padded lanes' values are zeroed by a
// VALU select before packing.
__global__ __launch_bounds__(256) void k_matvec(const float* __restrict__ win,
                                                const float* __restrict__ wrec,
                                                WS* ws) {
  __shared__ int s_cols[2 * MAXC];
  __shared__ int s_cnt[2];

  const int c0 = (ws->cnt[0] > MAXC) ? MAXC : ws->cnt[0];
  const int c1 = (ws->cnt[1] > MAXC) ? MAXC : ws->cnt[1];
  for (int t = threadIdx.x; t < MAXC; t += 256) {
    s_cols[t]        = (t < c0) ? ws->cols_x[t] : 0;
    s_cols[MAXC + t] = (t < c1) ? ws->cols_y[t] : 0;
  }
  if (threadIdx.x == 0) { s_cnt[0] = c0; s_cnt[1] = c1; }
  __syncthreads();

  const int lane     = threadIdx.x & 31;
  const int wave     = threadIdx.x >> 5;
  const int row_base = blockIdx.x * 128 + wave * 16;
  const int row      = row_base + (lane & 15);
  const int khi      = (lane & 16) ? 8 : 0;   // ISA 16-bit A layout half-split

  V16 ones;
#pragma unroll
  for (int e = 0; e < 16; ++e) ones.e[e] = (__bf16)1.0f;

  v8f acc = {};

#pragma unroll
  for (int m = 0; m < 2; ++m) {
    const float* __restrict__ W = (m == 0) ? win : wrec;
    const float* __restrict__ Wrow = W + (size_t)row * NN;
    const int* cols = (m == 0) ? s_cols : (s_cols + MAXC);
    const int  cnt  = s_cnt[m];                 // uniform across grid
    const int  nch  = (cnt + 31) >> 5;          // uniform -> EXEC all-1 at WMMA
    for (int kc = 0; kc < nch; ++kc) {
      const int kb = kc * 32 + khi;
      // 16 K-indices for this lane = two contiguous 8-int runs (16B aligned)
      int4 ca = *(const int4*)&cols[kb];
      int4 cb = *(const int4*)&cols[kb + 4];
      int4 cc = *(const int4*)&cols[kb + 16];
      int4 cd = *(const int4*)&cols[kb + 20];
      // unconditional scattered gathers -> maximum outstanding loads
      float v[16];
      v[0]  = __builtin_nontemporal_load(Wrow + ca.x);
      v[1]  = __builtin_nontemporal_load(Wrow + ca.y);
      v[2]  = __builtin_nontemporal_load(Wrow + ca.z);
      v[3]  = __builtin_nontemporal_load(Wrow + ca.w);
      v[4]  = __builtin_nontemporal_load(Wrow + cb.x);
      v[5]  = __builtin_nontemporal_load(Wrow + cb.y);
      v[6]  = __builtin_nontemporal_load(Wrow + cb.z);
      v[7]  = __builtin_nontemporal_load(Wrow + cb.w);
      v[8]  = __builtin_nontemporal_load(Wrow + cc.x);
      v[9]  = __builtin_nontemporal_load(Wrow + cc.y);
      v[10] = __builtin_nontemporal_load(Wrow + cc.z);
      v[11] = __builtin_nontemporal_load(Wrow + cc.w);
      v[12] = __builtin_nontemporal_load(Wrow + cd.x);
      v[13] = __builtin_nontemporal_load(Wrow + cd.y);
      v[14] = __builtin_nontemporal_load(Wrow + cd.z);
      v[15] = __builtin_nontemporal_load(Wrow + cd.w);
      V16 a;
#pragma unroll
      for (int e = 0; e < 16; ++e) {
        int k = kb + ((e < 8) ? e : (e + 8));
        a.e[e] = (__bf16)((k < cnt) ? v[e] : 0.0f);  // mask padding
      }
      acc = __builtin_amdgcn_wmma_f32_16x16x32_bf16(
          false, a.v, false, ones.v, (short)0, acc, false, false);
    }
  }

  // D layout: VGPR r = M=r (lanes 0-15) / M=r+8 (lanes 16-31); all N columns
  // equal (B is ones) -> lane 0 holds rows 0..7, lane 16 rows 8..15.
  if ((lane & 15) == 0) {
    int rb = row_base + khi;
#pragma unroll
    for (int r = 0; r < 8; ++r) ws->yv[rb + r] = acc[r];
  }
}

// ---------------- histogram of integer activation counts ----------------
__global__ void k_hist(WS* ws) {
  int i = blockIdx.x * blockDim.x + threadIdx.x;
  if (i >= NN) return;
  int v = (int)(ws->yv[i] + 0.5f);
  v = (v < 0) ? 0 : ((v > NBINS - 1) ? NBINS - 1 : v);
  atomicAdd(&ws->hist[v], 1);
}

// ---------------- threshold: suffix-sum over bins ----------------
__global__ void k_thresh(WS* ws) {
  __shared__ int lds[NBINS];
  int t = threadIdx.x;
  lds[t] = ws->hist[t];
  __syncthreads();
  // inclusive suffix sum S[b] = count(y >= b)
  for (int off = 1; off < NBINS; off <<= 1) {
    int add = (t + off < NBINS) ? lds[t + off] : 0;
    __syncthreads();
    lds[t] += add;
    __syncthreads();
  }
  int Sb  = lds[t];
  int Sb1 = (t == NBINS - 1) ? 0 : lds[t + 1];
  if (Sb >= KACT && Sb1 < KACT) {   // exactly one bin satisfies this
    ws->thresh = t;
    ws->ntake  = KACT - Sb1;        // ties at t to take, lowest index first
  }
}

// ---------------- emit k-WTA output with index-ordered tie-break ----------------
__global__ void k_out(const WS* __restrict__ ws, float* __restrict__ out) {
  __shared__ int lds[1024];
  __shared__ int s_base;
  const int t     = ws->thresh;
  const int ntake = ws->ntake;
  if (threadIdx.x == 0) s_base = 0;
  __syncthreads();
  for (int p = 0; p < NN / 1024; ++p) {
    int i   = p * 1024 + threadIdx.x;
    int v   = (int)(ws->yv[i] + 0.5f);
    int tie = (v == t) ? 1 : 0;
    lds[threadIdx.x] = tie;
    __syncthreads();
    for (int off = 1; off < 1024; off <<= 1) {       // inclusive scan
      int add = (threadIdx.x >= off) ? lds[threadIdx.x - off] : 0;
      __syncthreads();
      lds[threadIdx.x] += add;
      __syncthreads();
    }
    int excl = lds[threadIdx.x] - tie;
    int base = s_base;
    out[i] = (v > t || (tie && (base + excl) < ntake)) ? 1.0f : 0.0f;
    __syncthreads();
    if (threadIdx.x == 1023) s_base = base + lds[1023];
    __syncthreads();
  }
}

extern "C" void kernel_launch(void* const* d_in, const int* in_sizes, int n_in,
                              void* d_out, int out_size, void* d_ws, size_t ws_size,
                              hipStream_t stream) {
  const float* x    = (const float*)d_in[0];
  const float* ylat = (const float*)d_in[1];
  const float* win  = (const float*)d_in[2];
  const float* wrec = (const float*)d_in[3];
  float* out = (float*)d_out;
  WS* ws = (WS*)d_ws;
  (void)in_sizes; (void)n_in; (void)out_size; (void)ws_size;

  k_init   <<<1,   1024, 0, stream>>>(ws);
  k_compact<<<16,  1024, 0, stream>>>(x, ylat, ws);
  k_matvec <<<128,  256, 0, stream>>>(win, wrec, ws);
  k_hist   <<<16,  1024, 0, stream>>>(ws);
  k_thresh <<<1,  NBINS, 0, stream>>>(ws);
  k_out    <<<1,   1024, 0, stream>>>(ws, out);
}
